// CausalSelfAttention_47957604827555
// MI455X (gfx1250) — matrix operand order repaired and checked by reading, not compiled, not run on hardware
//
#include <hip/hip_runtime.h>
#include <hip/hip_bf16.h>
#include <stdint.h>

// ---------------------------------------------------------------------------
// Types / helpers
// ---------------------------------------------------------------------------
typedef __attribute__((ext_vector_type(16))) _Float16 v16h;
typedef __attribute__((ext_vector_type(8)))  float    v8f;
typedef __attribute__((ext_vector_type(4)))  unsigned u32x4;
typedef __attribute__((ext_vector_type(4)))  int      v4i;

#define AS1 __attribute__((address_space(1)))
#define AS3 __attribute__((address_space(3)))

#if __has_builtin(__builtin_amdgcn_global_load_async_to_lds_b128) && \
    __has_builtin(__builtin_amdgcn_s_wait_asynccnt)
#define HAVE_ASYNC_LDS 1
#else
#define HAVE_ASYNC_LDS 0
#endif

union FragU {
    unsigned u[8];
    u32x4    q[2];
    v16h     h;
};

static __device__ inline unsigned pack_h2(float a, float b) {
    union { _Float16 h[2]; unsigned u; } p;
    p.h[0] = (_Float16)a;
    p.h[1] = (_Float16)b;
    return p.u;
}

// ---------------------------------------------------------------------------
// f32 -> f16 conversion (one pass; makes all GEMM operands 16-bit so tile
// staging is a raw async copy and steady-state HBM traffic is halved).
// ---------------------------------------------------------------------------
__global__ __launch_bounds__(256) void cvt_f32_f16(
    const float* __restrict__ s, _Float16* __restrict__ d, long long n)
{
    long long i = ((long long)blockIdx.x * 256 + threadIdx.x) * 4;
    if (i + 3 < n) {
        const float* sp = s + i;
        unsigned* dp = (unsigned*)(d + i);
        dp[0] = pack_h2(sp[0], sp[1]);
        dp[1] = pack_h2(sp[2], sp[3]);
    }
}

// ---------------------------------------------------------------------------
// Tiled GEMM:  C[M,N] = A[M,K] * B[K,N]; A,B f16 in memory, f16 WMMA compute,
// f32 accumulate, CT output. Block tile 64(M) x 128(N), K-step 32.
// 256 threads = 8 waves as 2x4; each wave computes a 32x32 tile (2x2 wmma).
// A tile staged with GLOBAL_LOAD_ASYNC_TO_LDS_B128 (ASYNCcnt path) when the
// toolchain exposes it; B tile staged transposed (pair-packed) for B-frags.
// ---------------------------------------------------------------------------
#define BM 64
#define BN 128
#define BK 32

template <typename CT>
__global__ __launch_bounds__(256) void gemm_h(
    const _Float16* __restrict__ A, const _Float16* __restrict__ Bm,
    CT* __restrict__ C, int M, int N, int K)
{
    __shared__ __align__(16) _Float16 lA[BM][BK];        // row-major [m][k], 4KB
    __shared__ __align__(16) unsigned lB[BN][BK / 2];    // transposed pairs, 8KB

    const int tid  = threadIdx.x;
    const int lane = tid & 31;
    const int wave = tid >> 5;
    const int wM = wave >> 2;        // 0..1
    const int wN = wave & 3;         // 0..3
    const int lr = lane & 15;
    const int hi = lane >> 4;

    const int rowBase = blockIdx.y * BM;
    const int colBase = blockIdx.x * BN;

    v8f acc[2][2] = {};

    for (int k0 = 0; k0 < K; k0 += BK) {
        // ---- stage A tile: 64x32 f16 = 4KB = 256 threads x 16B ----
        {
            int r  = tid >> 2;           // 0..63
            int c4 = (tid & 3) * 8;      // half index 0,8,16,24
            const _Float16* src = A + (size_t)(rowBase + r) * K + k0 + c4;
#if HAVE_ASYNC_LDS
            __builtin_amdgcn_global_load_async_to_lds_b128(
                (AS1 v4i*)src, (AS3 v4i*)&lA[r][c4], 0, 0);
#else
            *(u32x4*)&lA[r][c4] = *(const u32x4*)src;
#endif
        }
        // ---- stage B tile transposed: 32x128 -> [n][kpair] (2048 pairs) ----
        #pragma unroll
        for (int i = 0; i < 8; ++i) {
            int p  = tid + 256 * i;      // 0..2047
            int n  = p >> 4;             // 0..127
            int kp = p & 15;
            const _Float16* s = Bm + (size_t)(k0 + 2 * kp) * N + colBase + n;
            union { _Float16 h[2]; unsigned u; } pk;
            pk.h[0] = s[0];
            pk.h[1] = s[N];
            lB[n][kp] = pk.u;
        }
        // prefetch next K tile while we compute this one
        if (k0 + BK < K) {
            __builtin_prefetch(A + (size_t)(rowBase + (tid & 63)) * K + k0 + BK, 0, 1);
            __builtin_prefetch(Bm + (size_t)(k0 + BK + lane) * N + colBase + ((tid >> 5) << 4), 0, 1);
        }
#if HAVE_ASYNC_LDS
        __builtin_amdgcn_s_wait_asynccnt(0);
#endif
        __syncthreads();

        FragU a[2], b[2];
        #pragma unroll
        for (int mt = 0; mt < 2; ++mt) {
            const _Float16* pA = &lA[32 * wM + 16 * mt + lr][0];
            a[mt].q[0] = *(const u32x4*)&pA[8 * hi];         // K = 8*hi + 0..7
            a[mt].q[1] = *(const u32x4*)&pA[16 + 8 * hi];    // K = 16 + 8*hi + 0..7
        }
        #pragma unroll
        for (int nt = 0; nt < 2; ++nt) {
            const unsigned* pB = &lB[32 * wN + 16 * nt + lr][0];
            b[nt].q[0] = *(const u32x4*)&pB[4 * hi];         // K = 8*hi + 0..7
            b[nt].q[1] = *(const u32x4*)&pB[8 + 4 * hi];     // K = 16 + 8*hi + 0..7
        }
        #pragma unroll
        for (int mt = 0; mt < 2; ++mt)
            #pragma unroll
            for (int nt = 0; nt < 2; ++nt)
                acc[mt][nt] = __builtin_amdgcn_wmma_f32_16x16x32_f16(
                    false, a[mt].h, false, b[nt].h, (short)0, acc[mt][nt], false, false);
        __syncthreads();
    }

    // ---- write back (C/D layout: vgpr r -> row r + 8*hi, col = lr) ----
    #pragma unroll
    for (int mt = 0; mt < 2; ++mt)
        #pragma unroll
        for (int nt = 0; nt < 2; ++nt)
            #pragma unroll
            for (int r = 0; r < 8; ++r) {
                int row = rowBase + 32 * wM + 16 * mt + 8 * hi + r;
                int col = colBase + 32 * wN + 16 * nt + lr;
                C[(size_t)row * N + col] = (CT)acc[mt][nt][r];
            }
}

// ---------------------------------------------------------------------------
// Fused: gate = 2*sigmoid(x[:,:32] @ Wg); v += gate*ve; RoPE(q,k); RMSNorm(q,k)
// Raw q/k/v are f16 (from the WMMA GEMMs); math in f32. Writes q,k f16
// row-major [bh][t][d]; v transposed [b,kv][d][t]. 1/sqrt(128) folded into q.
// One block per (b,t); 256 threads; 16 threads per head.
// ---------------------------------------------------------------------------
__global__ __launch_bounds__(256) void qkv_post(
    const float* __restrict__ x, const float* __restrict__ ve,
    const float* __restrict__ cosT, const float* __restrict__ sinT,
    const float* __restrict__ Wg,
    const _Float16* __restrict__ q_raw, const _Float16* __restrict__ k_raw,
    const _Float16* __restrict__ v_raw,
    _Float16* __restrict__ qh, _Float16* __restrict__ kh, _Float16* __restrict__ vT,
    int T)
{
    const int bt = blockIdx.x;
    const int b = bt / T, t = bt - b * T;
    __shared__ float sc[64], ss[64], sg[4];
    const int tid = threadIdx.x;

    if (tid < 64) {
        sc[tid] = cosT[(size_t)t * 64 + tid];
        ss[tid] = sinT[(size_t)t * 64 + tid];
    }
    if (tid < 4) {
        float g = 0.f;
        const float* xrow = x + (size_t)bt * 2048;
        #pragma unroll
        for (int c = 0; c < 32; ++c) g += xrow[c] * Wg[c * 4 + tid];
        sg[tid] = 2.f / (1.f + __expf(-g));
    }
    __syncthreads();

    const float inv_sqrt_d = 0.088388347648318447f;  // 1/sqrt(128)
    const int grp = tid >> 4, ln = tid & 15;

    // ---- Q: 16 heads, one per 16-thread group; each thread: 4 rope pairs ----
    {
        const _Float16* src = q_raw + (size_t)bt * 2048 + grp * 128;
        float r1[4], r2[4], ssq = 0.f;
        #pragma unroll
        for (int j = 0; j < 4; ++j) {
            int d = ln * 4 + j;
            float x1 = (float)src[d], x2 = (float)src[d + 64];
            float c = sc[d], s = ss[d];
            r1[j] = x1 * c + x2 * s;
            r2[j] = x2 * c - x1 * s;
            ssq += r1[j] * r1[j] + r2[j] * r2[j];
        }
        #pragma unroll
        for (int m = 1; m < 16; m <<= 1) ssq += __shfl_xor(ssq, m, 32);
        float scale = rsqrtf(ssq * (1.f / 128.f) + 1e-6f) * inv_sqrt_d;
        _Float16* dst = qh + (((size_t)b * 16 + grp) * T + t) * 128;
        #pragma unroll
        for (int j = 0; j < 4; ++j) {
            int d = ln * 4 + j;
            dst[d]      = (_Float16)(r1[j] * scale);
            dst[d + 64] = (_Float16)(r2[j] * scale);
        }
    }
    // ---- K: 4 KV heads (threads 0..63 = waves 0,1 fully active) ----
    if (grp < 4) {
        const _Float16* src = k_raw + (size_t)bt * 512 + grp * 128;
        float r1[4], r2[4], ssq = 0.f;
        #pragma unroll
        for (int j = 0; j < 4; ++j) {
            int d = ln * 4 + j;
            float x1 = (float)src[d], x2 = (float)src[d + 64];
            float c = sc[d], s = ss[d];
            r1[j] = x1 * c + x2 * s;
            r2[j] = x2 * c - x1 * s;
            ssq += r1[j] * r1[j] + r2[j] * r2[j];
        }
        #pragma unroll
        for (int m = 1; m < 16; m <<= 1) ssq += __shfl_xor(ssq, m, 32);
        float scale = rsqrtf(ssq * (1.f / 128.f) + 1e-6f);
        _Float16* dst = kh + (((size_t)b * 4 + grp) * T + t) * 128;
        #pragma unroll
        for (int j = 0; j < 4; ++j) {
            int d = ln * 4 + j;
            dst[d]      = (_Float16)(r1[j] * scale);
            dst[d + 64] = (_Float16)(r2[j] * scale);
        }
    }
    // ---- V: v + gate*ve, stored transposed [d][t] ----
    #pragma unroll
    for (int i = 0; i < 2; ++i) {
        int idx = tid + 256 * i;         // 0..511
        int kv = idx >> 7, d = idx & 127;
        float val = (float)v_raw[(size_t)bt * 512 + idx] + sg[kv] * ve[(size_t)bt * 512 + idx];
        vT[(((size_t)b * 4 + kv) * 128 + d) * T + t] = (_Float16)val;
    }
}

// ---------------------------------------------------------------------------
// Windowed attention, flash style. One wave per (b, h, 16-row q tile).
// Reference mask: valid iff 0 <= s - t < window. Output written in f16 for
// the final WMMA GEMM.
// ---------------------------------------------------------------------------
__global__ __launch_bounds__(32) void attn_fa(
    const _Float16* __restrict__ qh, const _Float16* __restrict__ kh,
    const _Float16* __restrict__ vT, const int* __restrict__ winp,
    _Float16* __restrict__ yout, int T)
{
    __shared__ __align__(16) _Float16 pbuf[16][32];

    const int lane = threadIdx.x;
    const int lr = lane & 15, hi = lane >> 4;
    const int t0 = blockIdx.x * 16;
    const int h  = blockIdx.y;
    const int b  = blockIdx.z;
    const int kv = h >> 2;                 // GQA: jnp.repeat -> h // 4
    const int win = *winp;

    // Q fragments (A layout), 4 K-chunks of 32 covering d=0..127
    FragU qf[4];
    const _Float16* qrow = qh + (((size_t)b * 16 + h) * T + t0 + lr) * 128;
    #pragma unroll
    for (int c = 0; c < 4; ++c) {
        qf[c].q[0] = *(const u32x4*)&qrow[32 * c + 8 * hi];
        qf[c].q[1] = *(const u32x4*)&qrow[32 * c + 16 + 8 * hi];
    }

    v8f y[8] = {};
    float m8[8], l8[8];
    #pragma unroll
    for (int r = 0; r < 8; ++r) { m8[r] = -1e30f; l8[r] = 0.f; }

    const _Float16* kbase = kh + ((size_t)b * 4 + kv) * T * 128;
    const _Float16* vbase = vT + ((size_t)b * 4 + kv) * 128 * (size_t)T;

    int send = t0 + 15 + win;
    if (send > T) send = T;

    for (int sc0 = t0 & ~31; sc0 < send; sc0 += 32) {
        // ---- scores S[16 x 32]: two 16x16 tiles, K=128 via 4 wmmas each ----
        v8f sacc[2];
        #pragma unroll
        for (int st = 0; st < 2; ++st) {
            v8f a = {};
            const _Float16* krow = kbase + (size_t)(sc0 + 16 * st + lr) * 128;
            #pragma unroll
            for (int c = 0; c < 4; ++c) {
                FragU kf;
                kf.q[0] = *(const u32x4*)&krow[32 * c + 8 * hi];
                kf.q[1] = *(const u32x4*)&krow[32 * c + 16 + 8 * hi];
                a = __builtin_amdgcn_wmma_f32_16x16x32_f16(
                        false, qf[c].h, false, kf.h, (short)0, a, false, false);
            }
            sacc[st] = a;
        }

        // ---- masked online softmax; C layout: vgpr r -> row r+8*hi, col lr ----
        float corr[8];
        #pragma unroll
        for (int r = 0; r < 8; ++r) {
            int t = t0 + r + 8 * hi;
            int s0 = sc0 + lr, s1 = s0 + 16;
            float v0 = sacc[0][r], v1 = sacc[1][r];
            int d0 = s0 - t, d1 = s1 - t;
            if (d0 < 0 || d0 >= win) v0 = -__builtin_inff();
            if (d1 < 0 || d1 >= win) v1 = -__builtin_inff();
            float mx = fmaxf(v0, v1);
            #pragma unroll
            for (int m = 1; m < 16; m <<= 1) mx = fmaxf(mx, __shfl_xor(mx, m, 32));
            float mn = fmaxf(m8[r], mx);
            float c  = __expf(m8[r] - mn);
            float p0 = __expf(v0 - mn);
            float p1 = __expf(v1 - mn);
            float rs = p0 + p1;
            #pragma unroll
            for (int m = 1; m < 16; m <<= 1) rs += __shfl_xor(rs, m, 32);
            l8[r] = l8[r] * c + rs;
            m8[r] = mn;
            corr[r] = c;
            pbuf[r + 8 * hi][lr]      = (_Float16)p0;
            pbuf[r + 8 * hi][16 + lr] = (_Float16)p1;
        }
        #pragma unroll
        for (int nt = 0; nt < 8; ++nt)
            #pragma unroll
            for (int r = 0; r < 8; ++r)
                y[nt][r] *= corr[r];
        __syncthreads();   // LDS transpose of P visible

        // ---- P fragment (A layout) + P@V: 8 wmmas covering d=0..127 ----
        FragU pf;
        pf.q[0] = *(const u32x4*)&pbuf[lr][8 * hi];
        pf.q[1] = *(const u32x4*)&pbuf[lr][16 + 8 * hi];
        #pragma unroll
        for (int nt = 0; nt < 8; ++nt) {
            const _Float16* vrow = vbase + (size_t)(16 * nt + lr) * T;
            FragU vf;
            vf.q[0] = *(const u32x4*)&vrow[sc0 + 8 * hi];
            vf.q[1] = *(const u32x4*)&vrow[sc0 + 16 + 8 * hi];
            y[nt] = __builtin_amdgcn_wmma_f32_16x16x32_f16(
                        false, pf.h, false, vf.h, (short)0, y[nt], false, false);
        }
        __syncthreads();   // pbuf reuse next chunk
    }

    // ---- finalize and write y[b,t,h*128+d] as f16 ----
    #pragma unroll
    for (int nt = 0; nt < 8; ++nt)
        #pragma unroll
        for (int r = 0; r < 8; ++r) {
            int t = t0 + r + 8 * hi;
            yout[((size_t)b * T + t) * 2048 + h * 128 + 16 * nt + lr] =
                (_Float16)(y[nt][r] / l8[r]);
        }
}

// ---------------------------------------------------------------------------
// Launch
// ---------------------------------------------------------------------------
extern "C" void kernel_launch(void* const* d_in, const int* in_sizes, int n_in,
                              void* d_out, int out_size, void* d_ws, size_t ws_size,
                              hipStream_t stream) {
    const float* x    = (const float*)d_in[0];
    const float* ve   = (const float*)d_in[1];
    const float* cosT = (const float*)d_in[2];
    const float* sinT = (const float*)d_in[3];
    const float* Wq   = (const float*)d_in[4];
    const float* Wk   = (const float*)d_in[5];
    const float* Wv   = (const float*)d_in[6];
    const float* Wg   = (const float*)d_in[7];
    const float* Wo   = (const float*)d_in[8];
    const int*   winp = (const int*)d_in[9];

    const int T = in_sizes[2] / 64;              // cos: (T, 1, 64)
    const int B = (int)(in_sizes[0] / ((size_t)T * 2048));
    const size_t M = (size_t)B * T;

    // workspace layout: all f16
    _Float16* p   = (_Float16*)d_ws;
    _Float16* xh  = p; p += M * 2048;
    _Float16* Wqh = p; p += (size_t)2048 * 2048;
    _Float16* Wkh = p; p += (size_t)2048 * 512;
    _Float16* Wvh = p; p += (size_t)2048 * 512;
    _Float16* Woh = p; p += (size_t)2048 * 2048;
    _Float16* qr  = p; p += M * 2048;
    _Float16* kr  = p; p += M * 512;
    _Float16* vr  = p; p += M * 512;
    _Float16* qhp = p; p += M * 2048;
    _Float16* khp = p; p += M * 512;
    _Float16* vTp = p; p += M * 512;
    _Float16* yh  = qr;   // alias: qr dead after qkv_post

    // one-shot f32 -> f16 conversions (4 elems/thread)
    cvt_f32_f16<<<(unsigned)(M * 2048 / 1024), 256, 0, stream>>>(x,  xh,  (long long)(M * 2048));
    cvt_f32_f16<<<(unsigned)(2048u * 2048u / 1024u), 256, 0, stream>>>(Wq, Wqh, 2048LL * 2048);
    cvt_f32_f16<<<(unsigned)(2048u * 512u  / 1024u), 256, 0, stream>>>(Wk, Wkh, 2048LL * 512);
    cvt_f32_f16<<<(unsigned)(2048u * 512u  / 1024u), 256, 0, stream>>>(Wv, Wvh, 2048LL * 512);
    cvt_f32_f16<<<(unsigned)(2048u * 2048u / 1024u), 256, 0, stream>>>(Wo, Woh, 2048LL * 2048);

    dim3 blk(256);
    gemm_h<_Float16><<<dim3(2048 / BN, (unsigned)(M / BM)), blk, 0, stream>>>(xh, Wqh, qr, (int)M, 2048, 2048);
    gemm_h<_Float16><<<dim3(512  / BN, (unsigned)(M / BM)), blk, 0, stream>>>(xh, Wkh, kr, (int)M, 512, 2048);
    gemm_h<_Float16><<<dim3(512  / BN, (unsigned)(M / BM)), blk, 0, stream>>>(xh, Wvh, vr, (int)M, 512, 2048);
    qkv_post<<<(unsigned)M, 256, 0, stream>>>(x, ve, cosT, sinT, Wg, qr, kr, vr, qhp, khp, vTp, T);
    attn_fa<<<dim3(T / 16, 16, B), 32, 0, stream>>>(qhp, khp, vTp, winp, yh, T);
    gemm_h<float><<<dim3(2048 / BN, (unsigned)(M / BM)), blk, 0, stream>>>(yh, Woh, (float*)d_out, (int)M, 2048, 2048);
}
